// Model_50981261804324
// MI455X (gfx1250) — matrix operand order repaired and checked by reading, not compiled, not run on hardware
//
#include <hip/hip_runtime.h>
#include <hip/hip_bf16.h>
#include <math.h>

typedef __attribute__((ext_vector_type(16))) __bf16 bf16x16;
typedef __attribute__((ext_vector_type(8)))  float  f32x8;
typedef unsigned short u16;

static_assert(sizeof(bf16x16) == 32, "bf16x16 must be 32 bytes");
static_assert(sizeof(f32x8) == 32, "f32x8 must be 32 bytes");

// ---------------- helpers ----------------
__device__ __forceinline__ u16 f2bf(float f) {
  unsigned u = __float_as_uint(f);
  u += 0x7FFFu + ((u >> 16) & 1u);       // round-to-nearest-even
  return (u16)(u >> 16);
}
__device__ __forceinline__ float bf2f(u16 h) {
  return __uint_as_float(((unsigned)h) << 16);
}
__device__ __forceinline__ float gelu_tanh(float x) {
  return 0.5f * x * (1.f + tanhf(0.7978845608f * (x + 0.044715f * x * x * x)));
}
__device__ __forceinline__ bf16x16 frag16(const u16* p0, const u16* p1) {
  union U { bf16x16 v; uint4 q[2]; } u;
  u.q[0] = *(const uint4*)p0;
  u.q[1] = *(const uint4*)p1;
  return u.v;
}

// ---------------- bf16 WMMA GEMM ----------------
// C[M,N] = act( A[M,K](bf16,row-major,lda) * Bt[N,K](bf16,row-major,ldb)^T + bias[n] )
// Batched via blockIdx.z with two-level (zb,zh) strides: off = zb*s1 + zh*s2.
// Block tile 128x128; 8 waves (4 in M x 2 in N); wave tile 32x64 = 2x4 WMMA tiles.
// Double-buffered LDS + register prefetch: one barrier per K-step of 32.
__global__ __launch_bounds__(256) void k_gemm(
    const u16* __restrict__ A, int lda, long aS1, long aS2,
    const u16* __restrict__ Bt, int ldb, long bS1, long bS2,
    const float* __restrict__ bias,
    u16* __restrict__ C, int ldc, long cS1, long cS2,
    int M, int N, int K, int batchH, int act)
{
  __shared__ u16 As[2][128][40];   // 32 K-cols + pad (row stride 80B, 16B-aligned)
  __shared__ u16 Bs[2][128][40];

  const int z  = blockIdx.z;
  const int zb = z / batchH;
  const int zh = z % batchH;
  A  += zb * aS1 + zh * aS2;
  Bt += zb * bS1 + zh * bS2;
  C  += zb * cS1 + zh * cS2;

  const int m0 = blockIdx.y * 128;
  const int n0 = blockIdx.x * 128;

  const int tid    = threadIdx.x;
  const int lane   = tid & 31;
  const int wave   = tid >> 5;
  const int wm     = (wave & 3) * 32;   // wave row offset within block tile
  const int wn     = (wave >> 2) * 64;  // wave col offset
  const int laneLo = lane & 15;
  const int kg     = lane >> 4;

  f32x8 acc[2][4];
  for (int i = 0; i < 2; ++i)
    for (int j = 0; j < 4; ++j)
      acc[i][j] = f32x8{};

  const int sr   = tid >> 2;       // 0..63 staging row (two passes: sr, sr+64)
  const int scol = (tid & 3) * 8;  // 0,8,16,24

  const uint4 z4 = {0u, 0u, 0u, 0u};
  uint4 a0, a1, b0, b1;
  const int gm0 = m0 + sr, gm1 = m0 + sr + 64;
  const int gn0 = n0 + sr, gn1 = n0 + sr + 64;

  auto loadAB = [&](int k0) {
    a0 = z4; a1 = z4; b0 = z4; b1 = z4;
    if (gm0 < M) a0 = *(const uint4*)(A + (long)gm0 * lda + k0 + scol);
    if (gm1 < M) a1 = *(const uint4*)(A + (long)gm1 * lda + k0 + scol);
    if (gn0 < N) b0 = *(const uint4*)(Bt + (long)gn0 * ldb + k0 + scol);
    if (gn1 < N) b1 = *(const uint4*)(Bt + (long)gn1 * ldb + k0 + scol);
  };
  auto storeAB = [&](int buf) {
    *(uint4*)&As[buf][sr][scol]      = a0;
    *(uint4*)&As[buf][sr + 64][scol] = a1;
    *(uint4*)&Bs[buf][sr][scol]      = b0;
    *(uint4*)&Bs[buf][sr + 64][scol] = b1;
  };

  loadAB(0);
  storeAB(0);
  __syncthreads();

  int cur = 0;
  for (int k0 = 0; k0 < K; k0 += 32) {
    const bool more = (k0 + 32) < K;
    if (more) {
      loadAB(k0 + 32);
      if (k0 + 64 < K) {   // gfx1250 global_prefetch for the chunk after next
        __builtin_prefetch(A + (long)gm0 * lda + k0 + 64 + scol, 0, 1);
        __builtin_prefetch(Bt + (long)gn0 * ldb + k0 + 64 + scol, 0, 1);
      }
    }

    // A fragment: lane m = lane&15; elems 0..7 -> K = kg*8..; elems 8..15 -> 16+kg*8..
    bf16x16 af[2], bfr[4];
    for (int i = 0; i < 2; ++i) {
      const u16* pa = &As[cur][wm + i * 16 + laneLo][kg * 8];
      af[i] = frag16(pa, pa + 16);
    }
    // B fragment: lane n = lane&15; elems 0..7 -> K = kg*8..; elems 8..15 -> 16+kg*8..
    for (int j = 0; j < 4; ++j) {
      const u16* pb = &Bs[cur][wn + j * 16 + laneLo][kg * 8];
      bfr[j] = frag16(pb, pb + 16);
    }
    for (int i = 0; i < 2; ++i)
      for (int j = 0; j < 4; ++j)
        acc[i][j] = __builtin_amdgcn_wmma_f32_16x16x32_bf16(
            false, af[i], false, bfr[j], (short)0, acc[i][j], false, false);

    if (more) {
      storeAB(cur ^ 1);
      __syncthreads();
      cur ^= 1;
    }
  }

  // epilogue: D element r -> m = r + (lane>=16 ? 8 : 0), n = lane&15
  const int mhi = (lane >> 4) * 8;
  for (int i = 0; i < 2; ++i) {
    for (int j = 0; j < 4; ++j) {
      int n = n0 + wn + j * 16 + laneLo;
      if (n >= N) continue;
      float bv = bias ? bias[n] : 0.f;
      for (int r = 0; r < 8; ++r) {
        int m = m0 + wm + i * 16 + r + mhi;
        if (m >= M) continue;
        float v = acc[i][j][r] + bv;
        if (act == 1) v = gelu_tanh(v);
        C[(long)m * ldc + n] = f2bf(v);
      }
    }
  }
}

// ---------------- data movement / small kernels ----------------

// build masked, flattened, K-padded input x: [8192, 928] bf16
__global__ __launch_bounds__(256) void k_prepx(const float* __restrict__ ccd,
                                               const float* __restrict__ amask,
                                               u16* __restrict__ x) {
  long i = (long)blockIdx.x * 256 + threadIdx.x;
  const long total = 8192L * 928;
  if (i >= total) return;
  long row = i / 928; int c = (int)(i % 928);
  float v = 0.f;
  if (c < 900) v = ccd[row * 900 + c] * amask[row];
  x[i] = f2bf(v);
}

// transpose-convert: src f32 [K,N] -> dst bf16 [N,Kp] (zero pad K..Kp)
__global__ __launch_bounds__(256) void k_convt(const float* __restrict__ src,
                                               u16* __restrict__ dst,
                                               int K, int N, int Kp) {
  long i = (long)blockIdx.x * 256 + threadIdx.x;
  long total = (long)N * Kp;
  if (i >= total) return;
  int n = (int)(i / Kp); int k = (int)(i % Kp);
  float v = (k < K) ? src[(long)k * N + n] : 0.f;
  dst[i] = f2bf(v);
}

// dkp[b,j] = DKP[b,:] @ W_gate[:,j]   (16x512, K=1024)
__global__ __launch_bounds__(256) void k_dkp(const float* __restrict__ DKP,
                                             const float* __restrict__ Wg,
                                             float* __restrict__ dkp) {
  int i = blockIdx.x * 256 + threadIdx.x;
  if (i >= 16 * 512) return;
  int b = i / 512, j = i % 512;
  float s = 0.f;
  for (int k = 0; k < 1024; ++k) s += DKP[b * 1024 + k] * Wg[k * 512 + j];
  dkp[i] = s;
}

// g[b,j] = sigmoid(dkp[b,:] @ Wg[:,j] + bg[j]) * km[b,j]   (16x3072, K=512)
__global__ __launch_bounds__(256) void k_gate(const float* __restrict__ dkp,
                                              const float* __restrict__ Wg,
                                              const float* __restrict__ bg,
                                              const int* __restrict__ cmask,
                                              u16* __restrict__ g) {
  int i = blockIdx.x * 256 + threadIdx.x;
  if (i >= 16 * 3072) return;
  int b = i / 3072, j = i % 3072;
  float s = bg[j];
  for (int k = 0; k < 512; ++k) s += dkp[b * 512 + k] * Wg[k * 3072 + j];
  float sg = 1.f / (1.f + expf(-s));
  float km = (j < 20 * 64) ? (float)cmask[b * 20 + (j >> 6)] : 1.f;
  g[i] = f2bf(sg * km);
}

// u[(b*512+l), j] *= g[b, j]   over [8192,3072]
__global__ __launch_bounds__(256) void k_mulg(u16* __restrict__ u,
                                              const u16* __restrict__ g) {
  long i = (long)blockIdx.x * 256 + threadIdx.x;
  const long total = 8192L * 3072;
  if (i >= total) return;
  long row = i / 3072; int j = (int)(i % 3072);
  int b = (int)(row >> 9);
  u[i] = f2bf(bf2f(u[i]) * bf2f(g[b * 3072 + j]));
}

__global__ __launch_bounds__(256) void k_add(const u16* __restrict__ a,
                                             const u16* __restrict__ b,
                                             u16* __restrict__ c, long n) {
  long i = (long)blockIdx.x * 256 + threadIdx.x;
  if (i >= n) return;
  c[i] = f2bf(bf2f(a[i]) + bf2f(b[i]));
}

// out += sinusoidal PE, out is [16,512,768]
__global__ __launch_bounds__(256) void k_addpe(u16* __restrict__ out) {
  long i = (long)blockIdx.x * 256 + threadIdx.x;
  const long total = 8192L * 768;
  if (i >= total) return;
  int d = (int)(i % 768);
  int l = (int)((i / 768) & 511);
  float freq = expf((float)(d & ~1) * (-9.2103403720f / 768.f));
  float pe = (d & 1) ? cosf((float)l * freq) : sinf((float)l * freq);
  out[i] = f2bf(bf2f(out[i]) + pe);
}

// LayerNorm over last dim D, one block (256 threads) per row
__global__ __launch_bounds__(256) void k_ln(const u16* __restrict__ in,
                                            u16* __restrict__ outp, int D) {
  __shared__ float red[256];
  long row = blockIdx.x;
  const u16* p = in + row * D;
  float s = 0.f;
  for (int d = threadIdx.x; d < D; d += 256) s += bf2f(p[d]);
  red[threadIdx.x] = s; __syncthreads();
  for (int o = 128; o; o >>= 1) { if (threadIdx.x < o) red[threadIdx.x] += red[threadIdx.x + o]; __syncthreads(); }
  float mean = red[0] / D; __syncthreads();
  float v = 0.f;
  for (int d = threadIdx.x; d < D; d += 256) { float x = bf2f(p[d]) - mean; v += x * x; }
  red[threadIdx.x] = v; __syncthreads();
  for (int o = 128; o; o >>= 1) { if (threadIdx.x < o) red[threadIdx.x] += red[threadIdx.x + o]; __syncthreads(); }
  float rstd = rsqrtf(red[0] / D + 1e-5f);
  for (int d = threadIdx.x; d < D; d += 256)
    outp[row * D + d] = f2bf((bf2f(p[d]) - mean) * rstd);
}

// softmax over last dim 512 of scores [192,512,512], scale + key-mask
__global__ __launch_bounds__(256) void k_softmax(u16* __restrict__ sc,
                                                 const float* __restrict__ amask,
                                                 float scale) {
  __shared__ float red[256];
  long row = blockIdx.x;                 // 192*512 rows
  int z = (int)(row >> 9);
  int b = z / 12;
  u16* p = sc + row * 512;
  float v[2]; float mx = -3.0e38f;
  for (int t = 0; t < 2; ++t) {
    int c = threadIdx.x + t * 256;
    float x = bf2f(p[c]) * scale;
    if (amask[b * 512 + c] == 0.f) x = -1e9f;
    v[t] = x; mx = fmaxf(mx, x);
  }
  red[threadIdx.x] = mx; __syncthreads();
  for (int o = 128; o; o >>= 1) { if (threadIdx.x < o) red[threadIdx.x] = fmaxf(red[threadIdx.x], red[threadIdx.x + o]); __syncthreads(); }
  float m = red[0]; __syncthreads();
  float s = 0.f;
  for (int t = 0; t < 2; ++t) { v[t] = expf(v[t] - m); s += v[t]; }
  red[threadIdx.x] = s; __syncthreads();
  for (int o = 128; o; o >>= 1) { if (threadIdx.x < o) red[threadIdx.x] += red[threadIdx.x + o]; __syncthreads(); }
  float inv = 1.f / red[0];
  for (int t = 0; t < 2; ++t) p[threadIdx.x + t * 256] = f2bf(v[t] * inv);
}

// Vt[(b*12+h)*64 + hd][l] = V[(b*512+l)*768 + h*64 + hd]
__global__ __launch_bounds__(256) void k_vt(const u16* __restrict__ V,
                                            u16* __restrict__ Vt) {
  long i = (long)blockIdx.x * 256 + threadIdx.x;
  const long total = 192L * 64 * 512;
  if (i >= total) return;
  int l = (int)(i % 512);
  long r = i / 512;
  int hd = (int)(r % 64);
  int z = (int)(r / 64);
  int b = z / 12, h = z % 12;
  Vt[i] = V[((long)(b * 512 + l)) * 768 + h * 64 + hd];
}

// gather last valid token, LN, head; writes preds[16] then last[16,768] (f32)
__global__ __launch_bounds__(256) void k_final(const u16* __restrict__ out,
                                               const float* __restrict__ amask,
                                               const float* __restrict__ Wh,
                                               const float* __restrict__ bh,
                                               float* __restrict__ dout) {
  __shared__ float red[256];
  int b = blockIdx.x;
  float s = 0.f;
  for (int l = threadIdx.x; l < 512; l += 256) s += amask[b * 512 + l];
  red[threadIdx.x] = s; __syncthreads();
  for (int o = 128; o; o >>= 1) { if (threadIdx.x < o) red[threadIdx.x] += red[threadIdx.x + o]; __syncthreads(); }
  int last = (int)(red[0] + 0.5f) - 1;
  if (last < 0) last = 0;
  __syncthreads();
  const u16* p = out + ((long)(b * 512 + last)) * 768;

  float m = 0.f;
  for (int d = threadIdx.x; d < 768; d += 256) m += bf2f(p[d]);
  red[threadIdx.x] = m; __syncthreads();
  for (int o = 128; o; o >>= 1) { if (threadIdx.x < o) red[threadIdx.x] += red[threadIdx.x + o]; __syncthreads(); }
  float mean = red[0] / 768.f; __syncthreads();
  float v = 0.f;
  for (int d = threadIdx.x; d < 768; d += 256) { float x = bf2f(p[d]) - mean; v += x * x; }
  red[threadIdx.x] = v; __syncthreads();
  for (int o = 128; o; o >>= 1) { if (threadIdx.x < o) red[threadIdx.x] += red[threadIdx.x + o]; __syncthreads(); }
  float rstd = rsqrtf(red[0] / 768.f + 1e-5f); __syncthreads();

  float acc = 0.f;
  for (int d = threadIdx.x; d < 768; d += 256) {
    float y = (bf2f(p[d]) - mean) * rstd;
    dout[16 + b * 768 + d] = y;
    acc += y * Wh[d];
  }
  red[threadIdx.x] = acc; __syncthreads();
  for (int o = 128; o; o >>= 1) { if (threadIdx.x < o) red[threadIdx.x] += red[threadIdx.x + o]; __syncthreads(); }
  if (threadIdx.x == 0) dout[b] = red[0] + bh[0];
}

// ---------------- orchestration ----------------
static inline unsigned cdiv(long a, long b) { return (unsigned)((a + b - 1) / b); }

extern "C" void kernel_launch(void* const* d_in, const int* in_sizes, int n_in,
                              void* d_out, int out_size, void* d_ws, size_t ws_size,
                              hipStream_t stream) {
  (void)in_sizes; (void)n_in; (void)out_size; (void)ws_size;

  const float* ccd    = (const float*)d_in[0];
  const float* amask  = (const float*)d_in[1];
  const float* DKP    = (const float*)d_in[2];
  const int*   cmask  = (const int*)d_in[3];
  const float* Wgate  = (const float*)d_in[4];
  const float* We1    = (const float*)d_in[5];
  const float* be1    = (const float*)d_in[6];
  const float* Weg    = (const float*)d_in[7];
  const float* beg    = (const float*)d_in[8];
  const float* We2    = (const float*)d_in[9];
  const float* be2    = (const float*)d_in[10];
  const float* Wm1    = (const float*)d_in[11];
  const float* bm1    = (const float*)d_in[12];
  const float* Wmg    = (const float*)d_in[13];
  const float* bmg    = (const float*)d_in[14];
  const float* Wm2    = (const float*)d_in[15];
  const float* bm2    = (const float*)d_in[16];
  const float* Wq     = (const float*)d_in[17];
  const float* Wk     = (const float*)d_in[18];
  const float* Wv     = (const float*)d_in[19];
  const float* Wo     = (const float*)d_in[20];
  const float* bq     = (const float*)d_in[21];
  const float* bk     = (const float*)d_in[22];
  const float* bv     = (const float*)d_in[23];
  const float* bo     = (const float*)d_in[24];
  const float* Wf1    = (const float*)d_in[25];
  const float* bf1    = (const float*)d_in[26];
  const float* Wfg    = (const float*)d_in[27];
  const float* bfg    = (const float*)d_in[28];
  const float* Wf2    = (const float*)d_in[29];
  const float* bf2    = (const float*)d_in[30];
  const float* Wh     = (const float*)d_in[31];
  const float* bh     = (const float*)d_in[32];
  float* dout = (float*)d_out;

  // --- workspace carve ---
  size_t off = 0;
  auto carve = [&](size_t bytes) -> void* {
    off = (off + 255) & ~(size_t)255;
    void* p = (char*)d_ws + off;
    off += bytes;
    return p;
  };
  const long M = 8192;
  u16* xbf  = (u16*)carve((size_t)M * 928 * 2);
  u16* We1t = (u16*)carve((size_t)3072 * 928 * 2);
  u16* We2t = (u16*)carve((size_t)768 * 3072 * 2);
  u16* Wm1t = (u16*)carve((size_t)2 * 3072 * 768 * 2);
  u16* Wm2t = (u16*)carve((size_t)2 * 768 * 3072 * 2);
  u16* Wqt  = (u16*)carve((size_t)2 * 768 * 768 * 2);
  u16* Wkt  = (u16*)carve((size_t)2 * 768 * 768 * 2);
  u16* Wvt  = (u16*)carve((size_t)2 * 768 * 768 * 2);
  u16* Wot  = (u16*)carve((size_t)2 * 768 * 768 * 2);
  u16* Wf1t = (u16*)carve((size_t)2 * 3072 * 768 * 2);
  u16* Wf2t = (u16*)carve((size_t)2 * 768 * 3072 * 2);
  float* dkp = (float*)carve((size_t)16 * 512 * 4);
  u16* gE   = (u16*)carve((size_t)16 * 3072 * 2);
  u16* gM[2] = { (u16*)carve((size_t)16 * 3072 * 2), (u16*)carve((size_t)16 * 3072 * 2) };
  u16* gF[2] = { (u16*)carve((size_t)16 * 3072 * 2), (u16*)carve((size_t)16 * 3072 * 2) };
  u16* ubuf = (u16*)carve((size_t)M * 3072 * 2);
  u16* xout = (u16*)carve((size_t)M * 768 * 2);
  u16* tnorm= (u16*)carve((size_t)M * 768 * 2);
  u16* tmp  = (u16*)carve((size_t)M * 768 * 2);
  u16* Qb   = (u16*)carve((size_t)M * 768 * 2);
  u16* Kb   = (u16*)carve((size_t)M * 768 * 2);
  u16* Vb   = (u16*)carve((size_t)M * 768 * 2);
  u16* Vt   = (u16*)carve((size_t)192 * 64 * 512 * 2);
  u16* Ob   = (u16*)carve((size_t)M * 768 * 2);
  u16* sc   = (u16*)carve((size_t)192 * 512 * 512 * 2);

  const dim3 T(256);
  auto gemm = [&](const u16* A, int lda, long aS1, long aS2,
                  const u16* Bt, int ldb, long bS1, long bS2,
                  const float* bias, u16* C, int ldc, long cS1, long cS2,
                  int Mm, int Nn, int Kk, int batches, int batchH, int act) {
    dim3 g(cdiv(Nn, 128), cdiv(Mm, 128), batches);
    k_gemm<<<g, T, 0, stream>>>(A, lda, aS1, aS2, Bt, ldb, bS1, bS2, bias,
                                C, ldc, cS1, cS2, Mm, Nn, Kk, batchH, act);
  };

  // --- weight conversions (transposed to [N,Kp] bf16) ---
  k_prepx<<<cdiv(M * 928, 256), T, 0, stream>>>(ccd, amask, xbf);
  k_convt<<<cdiv(3072L * 928, 256), T, 0, stream>>>(We1, We1t, 900, 3072, 928);
  k_convt<<<cdiv(768L * 3072, 256), T, 0, stream>>>(We2, We2t, 3072, 768, 3072);
  for (int i = 0; i < 2; ++i) {
    k_convt<<<cdiv(3072L * 768, 256), T, 0, stream>>>(Wm1 + (size_t)i * 768 * 3072, Wm1t + (size_t)i * 3072 * 768, 768, 3072, 768);
    k_convt<<<cdiv(768L * 3072, 256), T, 0, stream>>>(Wm2 + (size_t)i * 3072 * 768, Wm2t + (size_t)i * 768 * 3072, 3072, 768, 3072);
    k_convt<<<cdiv(768L * 768, 256), T, 0, stream>>>(Wq + (size_t)i * 768 * 768, Wqt + (size_t)i * 768 * 768, 768, 768, 768);
    k_convt<<<cdiv(768L * 768, 256), T, 0, stream>>>(Wk + (size_t)i * 768 * 768, Wkt + (size_t)i * 768 * 768, 768, 768, 768);
    k_convt<<<cdiv(768L * 768, 256), T, 0, stream>>>(Wv + (size_t)i * 768 * 768, Wvt + (size_t)i * 768 * 768, 768, 768, 768);
    k_convt<<<cdiv(768L * 768, 256), T, 0, stream>>>(Wo + (size_t)i * 768 * 768, Wot + (size_t)i * 768 * 768, 768, 768, 768);
    k_convt<<<cdiv(3072L * 768, 256), T, 0, stream>>>(Wf1 + (size_t)i * 768 * 3072, Wf1t + (size_t)i * 3072 * 768, 768, 3072, 768);
    k_convt<<<cdiv(768L * 3072, 256), T, 0, stream>>>(Wf2 + (size_t)i * 3072 * 768, Wf2t + (size_t)i * 768 * 3072, 3072, 768, 3072);
  }

  // --- gates ---
  k_dkp<<<cdiv(16L * 512, 256), T, 0, stream>>>(DKP, Wgate, dkp);
  k_gate<<<cdiv(16L * 3072, 256), T, 0, stream>>>(dkp, Weg, beg, cmask, gE);
  for (int i = 0; i < 2; ++i) {
    k_gate<<<cdiv(16L * 3072, 256), T, 0, stream>>>(dkp, Wmg + (size_t)i * 512 * 3072, bmg + i * 3072, cmask, gM[i]);
    k_gate<<<cdiv(16L * 3072, 256), T, 0, stream>>>(dkp, Wfg + (size_t)i * 512 * 3072, bfg + i * 3072, cmask, gF[i]);
  }

  // --- patch embed gated FFN ---
  gemm(xbf, 928, 0, 0, We1t, 928, 0, 0, be1, ubuf, 3072, 0, 0, (int)M, 3072, 928, 1, 1, 1);
  k_mulg<<<cdiv(M * 3072, 256), T, 0, stream>>>(ubuf, gE);
  gemm(ubuf, 3072, 0, 0, We2t, 3072, 0, 0, be2, xout, 768, 0, 0, (int)M, 768, 3072, 1, 1, 0);

  // --- intra MLP-MoE blocks ---
  for (int i = 0; i < 2; ++i) {
    k_ln<<<(unsigned)M, T, 0, stream>>>(xout, tnorm, 768);
    gemm(tnorm, 768, 0, 0, Wm1t + (size_t)i * 3072 * 768, 768, 0, 0, bm1 + i * 3072,
         ubuf, 3072, 0, 0, (int)M, 3072, 768, 1, 1, 1);
    k_mulg<<<cdiv(M * 3072, 256), T, 0, stream>>>(ubuf, gM[i]);
    gemm(ubuf, 3072, 0, 0, Wm2t + (size_t)i * 768 * 3072, 3072, 0, 0, bm2 + i * 768,
         tmp, 768, 0, 0, (int)M, 768, 3072, 1, 1, 0);
    k_add<<<cdiv(M * 768, 256), T, 0, stream>>>(xout, tmp, xout, M * 768);
  }

  // --- positional encoding ---
  k_addpe<<<cdiv(M * 768, 256), T, 0, stream>>>(xout);

  // --- encoder layers ---
  for (int i = 0; i < 2; ++i) {
    gemm(xout, 768, 0, 0, Wqt + (size_t)i * 768 * 768, 768, 0, 0, bq + i * 768, Qb, 768, 0, 0, (int)M, 768, 768, 1, 1, 0);
    gemm(xout, 768, 0, 0, Wkt + (size_t)i * 768 * 768, 768, 0, 0, bk + i * 768, Kb, 768, 0, 0, (int)M, 768, 768, 1, 1, 0);
    gemm(xout, 768, 0, 0, Wvt + (size_t)i * 768 * 768, 768, 0, 0, bv + i * 768, Vb, 768, 0, 0, (int)M, 768, 768, 1, 1, 0);
    // scores[b,h] = Q_bh [512,64] @ K_bh^T : Bt = K_bh [512,64]
    gemm(Qb, 768, 512L * 768, 64, Kb, 768, 512L * 768, 64, nullptr,
         sc, 512, 12L * 512 * 512, 512L * 512, 512, 512, 64, 192, 12, 0);
    k_softmax<<<192 * 512, T, 0, stream>>>(sc, amask, 0.125f);
    k_vt<<<cdiv(192L * 64 * 512, 256), T, 0, stream>>>(Vb, Vt);
    // o[b,h] = P_bh [512,512] @ V_bh [512,64] : Bt = Vt_bh [64,512]
    gemm(sc, 512, 12L * 512 * 512, 512L * 512, Vt, 512, 12L * 64 * 512, 64L * 512, nullptr,
         Ob, 768, 512L * 768, 64, 512, 64, 512, 192, 12, 0);
    gemm(Ob, 768, 0, 0, Wot + (size_t)i * 768 * 768, 768, 0, 0, bo + i * 768, tmp, 768, 0, 0, (int)M, 768, 768, 1, 1, 0);
    k_add<<<cdiv(M * 768, 256), T, 0, stream>>>(xout, tmp, tmp, M * 768);
    k_ln<<<(unsigned)M, T, 0, stream>>>(tmp, xout, 768);
    // gated FFN
    gemm(xout, 768, 0, 0, Wf1t + (size_t)i * 3072 * 768, 768, 0, 0, bf1 + i * 3072,
         ubuf, 3072, 0, 0, (int)M, 3072, 768, 1, 1, 1);
    k_mulg<<<cdiv(M * 3072, 256), T, 0, stream>>>(ubuf, gF[i]);
    gemm(ubuf, 3072, 0, 0, Wf2t + (size_t)i * 768 * 3072, 3072, 0, 0, bf2 + i * 768,
         tmp, 768, 0, 0, (int)M, 768, 3072, 1, 1, 0);
    k_add<<<cdiv(M * 768, 256), T, 0, stream>>>(xout, tmp, tmp, M * 768);
    k_ln<<<(unsigned)M, T, 0, stream>>>(tmp, xout, 768);
  }

  // --- head ---
  k_final<<<16, T, 0, stream>>>(xout, amask, Wh, bh, dout);
}
